// TwoWayTransformer_53420803228092
// MI455X (gfx1250) — matrix operand order repaired and checked by reading, not compiled, not run on hardware
//
#include <hip/hip_runtime.h>
#include <math.h>

// ---------------------------------------------------------------------------
// Problem constants (from the reference)
// ---------------------------------------------------------------------------
#define B_     16
#define EMBED  256
#define HW_    4096      // 64*64
#define NPTS   64
#define HEADS  8
#define MLPDIM 2048
#define LN_EPS 1e-5f

// ---------------------------------------------------------------------------
// CDNA5 WMMA types / helpers (wave32, 16x16x32 bf16 -> f32)
// ---------------------------------------------------------------------------
typedef __bf16 bf16_t;
typedef __attribute__((ext_vector_type(16))) __bf16 v16bf;
typedef __attribute__((ext_vector_type(8)))  __bf16 v8bf;
typedef __attribute__((ext_vector_type(8)))  float  v8f;

__device__ __forceinline__ bf16_t f2bf(float f) {
  unsigned int u = __builtin_bit_cast(unsigned int, f);
  unsigned int r = (u + 0x7FFFu + ((u >> 16) & 1u)) >> 16;   // round-nearest-even
  unsigned short s = (unsigned short)r;
  return __builtin_bit_cast(bf16_t, s);
}
__device__ __forceinline__ bf16_t bf_zero() {
  return __builtin_bit_cast(bf16_t, (unsigned short)0);
}
__device__ __forceinline__ unsigned int pack2(float a, float b) {
  unsigned int x = (unsigned int)__builtin_bit_cast(unsigned short, f2bf(a));
  unsigned int y = (unsigned int)__builtin_bit_cast(unsigned short, f2bf(b));
  return x | (y << 16);
}
__device__ __forceinline__ v16bf cat8(v8bf lo, v8bf hi) {
  return __builtin_shufflevector(lo, hi, 0, 1, 2, 3, 4, 5, 6, 7,
                                 8, 9, 10, 11, 12, 13, 14, 15);
}

__device__ __forceinline__ v8f wmma_bf16(v16bf a, v16bf b, v8f c) {
  // 8 args: (neg_a, A, neg_b, B, c_mod, C, reuse_a, reuse_b)
  return __builtin_amdgcn_wmma_f32_16x16x32_bf16(false, a, false, b,
                                                 (short)0, c, false, false);
}

// A-fragment (16x32 bf16, ISA 7.12.2): lane m = lane&15, half = lane>>4,
// element i holds K = half*8 + i + (i>=8 ? 8 : 0)  -> two contiguous 8-chunks.
// rowbase points at element [m][0] of the 16-row tile (stride already applied).
__device__ __forceinline__ v16bf load_a_frag(const bf16_t* rowbase, int ks, int half) {
  v8bf lo = *(const v8bf*)(rowbase + ks + half * 8);
  v8bf hi = *(const v8bf*)(rowbase + ks + 16 + half * 8);
  return cat8(lo, hi);
}
// B-fragment: lane n = lane&15, element i holds K = half*8 + i + (i>=8 ? 8 : 0).
// rowbase points at the transposed row [n][0] (K contiguous).
__device__ __forceinline__ v16bf load_b_frag(const bf16_t* rowbase, int ks, int half) {
  v8bf lo = *(const v8bf*)(rowbase + ks + half * 8);
  v8bf hi = *(const v8bf*)(rowbase + ks + 16 + half * 8);
  return cat8(lo, hi);
}

// ---------------------------------------------------------------------------
// GEMM: C[M,N] = act(A[M,K] @ W[K,N] + bias[N]) (+ residual)
// 256 threads = 8 waves; block tile 64x64, K-step 64 (2 WMMA K-steps).
// A tile staged row-major (bf16), W tile staged transposed so both fragment
// types are contiguous 16B LDS loads.
// ---------------------------------------------------------------------------
template <bool RELU, bool RESID>
__global__ __launch_bounds__(256)
void gemm_wmma(const float* __restrict__ A, const float* __restrict__ W,
               const float* __restrict__ bias, const float* __restrict__ resid,
               float* __restrict__ C, int M, int N, int K) {
  __shared__ bf16_t As[64][72];   // [m][k], 144B row stride (16B aligned)
  __shared__ bf16_t Bt[64][72];   // [n][k] transposed

  const int tid   = threadIdx.x;
  const int wave  = tid >> 5;
  const int lane  = tid & 31;
  const int half  = lane >> 4;
  const int l15   = lane & 15;
  const int ntile = wave & 3;          // 4 N-tiles of 16
  const int mt0   = (wave >> 2) * 2;   // each wave owns 2 stacked M-tiles

  const int blockM = blockIdx.y * 64;
  const int blockN = blockIdx.x * 64;

  v8f acc0 = {};
  v8f acc1 = {};

  for (int k0 = 0; k0 < K; k0 += 64) {
    if (k0 + 64 < K) {   // global_prefetch_b8 of next K tile
      __builtin_prefetch(A + (size_t)(blockM + (tid & 63)) * K + k0 + 64, 0, 1);
      __builtin_prefetch(W + (size_t)(k0 + 64 + (tid & 63)) * N + blockN, 0, 1);
    }
    // stage A: 64x64 f32 -> bf16 packed, ds_store_b64
#pragma unroll
    for (int t = 0; t < 4; ++t) {
      int idx = tid + t * 256;
      int r = idx >> 4, c4 = idx & 15;
      float4 v = *(const float4*)(A + (size_t)(blockM + r) * K + k0 + c4 * 4);
      uint2 p; p.x = pack2(v.x, v.y); p.y = pack2(v.z, v.w);
      *(uint2*)&As[r][c4 * 4] = p;
    }
    // stage W transposed: read [k][n] float4, scatter to Bt[n][k]
#pragma unroll
    for (int t = 0; t < 4; ++t) {
      int idx = tid + t * 256;
      int r = idx >> 4, c4 = idx & 15;   // r = k row, c4*4 = n base
      float4 v = *(const float4*)(W + (size_t)(k0 + r) * N + blockN + c4 * 4);
      Bt[c4 * 4 + 0][r] = f2bf(v.x);
      Bt[c4 * 4 + 1][r] = f2bf(v.y);
      Bt[c4 * 4 + 2][r] = f2bf(v.z);
      Bt[c4 * 4 + 3][r] = f2bf(v.w);
    }
    __syncthreads();

    const bf16_t* arow0 = &As[mt0 * 16 + l15][0];
    const bf16_t* arow1 = &As[(mt0 + 1) * 16 + l15][0];
    const bf16_t* brow  = &Bt[ntile * 16 + l15][0];
#pragma unroll
    for (int ks = 0; ks < 64; ks += 32) {
      v16bf bf = load_b_frag(brow,  ks, half);
      v16bf a0 = load_a_frag(arow0, ks, half);
      v16bf a1 = load_a_frag(arow1, ks, half);
      acc0 = wmma_bf16(a0, bf, acc0);
      acc1 = wmma_bf16(a1, bf, acc1);
    }
    __syncthreads();
  }

  // epilogue: C layout = 8 VGPRs, row m = r + 8*half, col n = lane&15
  const int gn = blockN + ntile * 16 + l15;
  const float bv = bias[gn];
#pragma unroll
  for (int t = 0; t < 2; ++t) {
    v8f acc = t ? acc1 : acc0;
    const int gm0 = blockM + (mt0 + t) * 16;
#pragma unroll
    for (int r = 0; r < 8; ++r) {
      int gm = gm0 + r + 8 * half;
      float v = acc[r] + bv;
      if (RELU)  v = fmaxf(v, 0.f);
      if (RESID) v += resid[(size_t)gm * N + gn];
      C[(size_t)gm * N + gn] = v;
    }
  }
}

// ---------------------------------------------------------------------------
// Flash attention, head dim D compile-time (16 or 32). One block = 4 waves =
// 64 q rows for one (batch, head). K/V tiles staged once per block into LDS;
// all fragments are contiguous 16B ds loads; pad regions zeroed once.
// ---------------------------------------------------------------------------
template <int D>
__global__ __launch_bounds__(128)
void attn_kernel(const float* __restrict__ Q, const float* __restrict__ Kp,
                 const float* __restrict__ Vp, float* __restrict__ O,
                 int nq, int nk, float scale) {
  constexpr int CINT = HEADS * D;
  constexpr int F4   = D / 4;            // float4 per row of one head
  const int b    = blockIdx.z;
  const int h    = blockIdx.y;
  const int q0   = blockIdx.x * 64;
  const int tid  = threadIdx.x;
  const int wave = tid >> 5;
  const int lane = tid & 31;
  const int half = lane >> 4;
  const int l15  = lane & 15;
  const int qrow0 = q0 + wave * 16;

  __shared__ float  Sbuf[4][16][17];
  __shared__ bf16_t Kt[16][40];          // [key][channel 0..31 + pad]
  __shared__ bf16_t Vt[32][40];          // [channel][key 0..31 + pad]
  __shared__ bf16_t Pbuf[4][16][32];     // [wave][qrow][key 0..31]
  __shared__ float  alphaBuf[4][16];
  __shared__ float  linvBuf[4][16];

  const bf16_t bz = bf_zero();

  // zero-fill once: pad channels/keys (cols >= D resp. >= 16) stay zero
  for (int i = tid; i < 16 * 40; i += 128) (&Kt[0][0])[i] = bz;
  for (int i = tid; i < 32 * 40; i += 128) (&Vt[0][0])[i] = bz;
  for (int i = tid; i < 4 * 16 * 32; i += 128) (&Pbuf[0][0][0])[i] = bz;

  // Q A-fragment: row = qrow0+l15, channels in two contiguous 8-chunks.
  v16bf qfrag;
  {
    const float* qb = Q + ((size_t)(b * nq + qrow0 + l15)) * CINT + h * D;
    float4 a0 = *(const float4*)(qb + half * 8);
    float4 a1 = *(const float4*)(qb + half * 8 + 4);
    qfrag[0] = f2bf(a0.x * scale); qfrag[1] = f2bf(a0.y * scale);
    qfrag[2] = f2bf(a0.z * scale); qfrag[3] = f2bf(a0.w * scale);
    qfrag[4] = f2bf(a1.x * scale); qfrag[5] = f2bf(a1.y * scale);
    qfrag[6] = f2bf(a1.z * scale); qfrag[7] = f2bf(a1.w * scale);
    if (D == 32) {
      float4 b0 = *(const float4*)(qb + 16 + half * 8);
      float4 b1 = *(const float4*)(qb + 16 + half * 8 + 4);
      qfrag[8]  = f2bf(b0.x * scale); qfrag[9]  = f2bf(b0.y * scale);
      qfrag[10] = f2bf(b0.z * scale); qfrag[11] = f2bf(b0.w * scale);
      qfrag[12] = f2bf(b1.x * scale); qfrag[13] = f2bf(b1.y * scale);
      qfrag[14] = f2bf(b1.z * scale); qfrag[15] = f2bf(b1.w * scale);
    } else {
#pragma unroll
      for (int i = 8; i < 16; ++i) qfrag[i] = bz;
    }
  }

  v8f oacc0 = {};
  v8f oacc1 = {};
  float mi = -1e30f, li = 0.f;    // valid in lanes 0..15 (one row each)

  for (int k0 = 0; k0 < nk; k0 += 16) {
    // ---- stage K tile (16 keys x D channels), packed b64 stores ----
    for (int idx = tid; idx < 16 * F4; idx += 128) {
      int r = idx / F4, c4 = idx % F4;
      float4 v = *(const float4*)(Kp + ((size_t)(b * nk + k0 + r)) * CINT +
                                  h * D + c4 * 4);
      uint2 p; p.x = pack2(v.x, v.y); p.y = pack2(v.z, v.w);
      *(uint2*)&Kt[r][c4 * 4] = p;
    }
    __syncthreads();   // B1: K tile visible (also fences prior-iter V reads)

    // ---- scores: S = (Q*scale) @ K^T via WMMA ----
    v16bf kfrag = load_b_frag(&Kt[l15][0], 0, half);
    v8f s = {};
    s = wmma_bf16(qfrag, kfrag, s);
#pragma unroll
    for (int r = 0; r < 8; ++r) Sbuf[wave][r + 8 * half][l15] = s[r];
    __syncthreads();   // B2

    // ---- online softmax (one row per lane 0..15) ----
    if (lane < 16) {
      float sv[16], rowmax = -1e30f;
#pragma unroll
      for (int j = 0; j < 16; ++j) {
        sv[j] = Sbuf[wave][l15][j];
        rowmax = fmaxf(rowmax, sv[j]);
      }
      float mnew  = fmaxf(mi, rowmax);
      float alpha = __expf(mi - mnew);
      float lsum  = 0.f;
      unsigned int* prow = (unsigned int*)&Pbuf[wave][l15][0];
#pragma unroll
      for (int j2 = 0; j2 < 8; ++j2) {
        float p0 = __expf(sv[2 * j2]     - mnew);
        float p1 = __expf(sv[2 * j2 + 1] - mnew);
        lsum += p0 + p1;
        prow[j2] = pack2(p0, p1);
      }
      li = li * alpha + lsum;
      mi = mnew;
      alphaBuf[wave][l15] = alpha;
      linvBuf[wave][l15]  = 1.f / li;
    }
    // ---- stage V tile transposed (channel-major) during softmax phase ----
    for (int idx = tid; idx < 16 * F4; idx += 128) {
      int r = idx / F4, c4 = idx % F4;   // r = key, c4*4 = channel base
      float4 v = *(const float4*)(Vp + ((size_t)(b * nk + k0 + r)) * CINT +
                                  h * D + c4 * 4);
      Vt[c4 * 4 + 0][r] = f2bf(v.x);
      Vt[c4 * 4 + 1][r] = f2bf(v.y);
      Vt[c4 * 4 + 2][r] = f2bf(v.z);
      Vt[c4 * 4 + 3][r] = f2bf(v.w);
    }
    __syncthreads();   // B3

    // ---- rescale and accumulate O += P @ V ----
#pragma unroll
    for (int r = 0; r < 8; ++r) {
      float a = alphaBuf[wave][r + 8 * half];
      oacc0[r] *= a;
      oacc1[r] *= a;
    }
    v16bf pfrag = load_a_frag(&Pbuf[wave][l15][0], 0, half);
    {
      v16bf vf0 = load_b_frag(&Vt[l15][0], 0, half);
      oacc0 = wmma_bf16(pfrag, vf0, oacc0);
    }
    if (D == 32) {
      v16bf vf1 = load_b_frag(&Vt[16 + l15][0], 0, half);
      oacc1 = wmma_bf16(pfrag, vf1, oacc1);
    }
  }

  __syncthreads();
#pragma unroll
  for (int r = 0; r < 8; ++r) {
    int m = r + 8 * half;
    float inv = linvBuf[wave][m];
    size_t base = ((size_t)(b * nq + qrow0 + m)) * CINT + h * D;
    O[base + l15] = oacc0[r] * inv;
    if (D == 32) O[base + 16 + l15] = oacc1[r] * inv;
  }
}

// ---------------------------------------------------------------------------
// LayerNorm over last dim (=256). One block per row, 256 threads.
// ---------------------------------------------------------------------------
__global__ __launch_bounds__(256)
void layernorm_kernel(const float* __restrict__ x, const float* __restrict__ g,
                      const float* __restrict__ bb, float* __restrict__ y) {
  const int C = 256;
  size_t row = blockIdx.x;
  int t = threadIdx.x;
  float v = x[row * C + t];

  __shared__ float red[256];
  red[t] = v;
  __syncthreads();
#pragma unroll
  for (int s = 128; s > 0; s >>= 1) {
    if (t < s) red[t] += red[t + s];
    __syncthreads();
  }
  float mu = red[0] / C;
  __syncthreads();
  float dv = v - mu;
  red[t] = dv * dv;
  __syncthreads();
#pragma unroll
  for (int s = 128; s > 0; s >>= 1) {
    if (t < s) red[t] += red[t + s];
    __syncthreads();
  }
  float var = red[0] / C;
  y[row * C + t] = dv * rsqrtf(var + LN_EPS) * g[t] + bb[t];
}

// ---------------------------------------------------------------------------
// Elementwise add, and BCHW -> B(HW)C transpose
// ---------------------------------------------------------------------------
__global__ void add_kernel(const float* __restrict__ a, const float* __restrict__ b,
                           float* __restrict__ c, size_t n) {
  size_t i = (size_t)blockIdx.x * blockDim.x + threadIdx.x;
  size_t st = (size_t)gridDim.x * blockDim.x;
  for (; i < n; i += st) c[i] = a[i] + b[i];
}

__global__ void bchw_to_bnc(const float* __restrict__ in, float* __restrict__ out) {
  size_t n = (size_t)B_ * EMBED * HW_;
  size_t i = (size_t)blockIdx.x * blockDim.x + threadIdx.x;
  size_t st = (size_t)gridDim.x * blockDim.x;
  for (; i < n; i += st) {
    size_t b = i / ((size_t)HW_ * EMBED);
    size_t rem = i % ((size_t)HW_ * EMBED);
    size_t p = rem / EMBED, c = rem % EMBED;
    out[i] = in[(b * EMBED + c) * HW_ + p];
  }
}

// ---------------------------------------------------------------------------
// Host-side orchestration
// ---------------------------------------------------------------------------
struct Lin  { const float* w; const float* b; };
struct Attn { Lin q, k, v, o; };
struct LN   { const float* g; const float* b; };

static void launch_gemm(const float* A, const float* W, const float* bias,
                        const float* resid, float* C, int M, int N, int K,
                        bool relu, hipStream_t s) {
  dim3 g(N / 64, M / 64), b(256);
  if (relu)
    gemm_wmma<true, false><<<g, b, 0, s>>>(A, W, bias, nullptr, C, M, N, K);
  else if (resid)
    gemm_wmma<false, true><<<g, b, 0, s>>>(A, W, bias, resid, C, M, N, K);
  else
    gemm_wmma<false, false><<<g, b, 0, s>>>(A, W, bias, nullptr, C, M, N, K);
}

static void launch_add(const float* a, const float* b, float* c, size_t n,
                       hipStream_t s) {
  size_t blocks = (n + 255) / 256;
  if (blocks > 8192) blocks = 8192;
  add_kernel<<<(unsigned)blocks, 256, 0, s>>>(a, b, c, n);
}

static void launch_ln(const LN& p, float* x, int rows, hipStream_t s) {
  layernorm_kernel<<<rows, 256, 0, s>>>(x, p.g, p.b, x);
}

// out = o_proj( attention(qin, kin, vin) ) [+ resid]
static void run_attention(const Attn& at, const float* qin, const float* kin,
                          const float* vin, int nq, int nk, int cint,
                          const float* resid, float* dst,
                          float* pq, float* pk, float* pv, float* abuf,
                          hipStream_t s) {
  const int Mq = B_ * nq, Mk = B_ * nk;
  const int d = cint / HEADS;
  const float scale = 1.f / sqrtf((float)d);
  launch_gemm(qin, at.q.w, at.q.b, nullptr, pq, Mq, cint, EMBED, false, s);
  launch_gemm(kin, at.k.w, at.k.b, nullptr, pk, Mk, cint, EMBED, false, s);
  launch_gemm(vin, at.v.w, at.v.b, nullptr, pv, Mk, cint, EMBED, false, s);
  dim3 g(nq / 64, HEADS, B_);
  if (d == 16)
    attn_kernel<16><<<g, 128, 0, s>>>(pq, pk, pv, abuf, nq, nk, scale);
  else
    attn_kernel<32><<<g, 128, 0, s>>>(pq, pk, pv, abuf, nq, nk, scale);
  launch_gemm(abuf, at.o.w, at.o.b, resid, dst, Mq, EMBED, cint, false, s);
}

extern "C" void kernel_launch(void* const* d_in, const int* in_sizes, int n_in,
                              void* d_out, int out_size, void* d_ws, size_t ws_size,
                              hipStream_t stream) {
  (void)in_sizes; (void)n_in; (void)out_size; (void)ws_size;
  auto F = [&](int i) { return (const float*)d_in[i]; };

  const float* image_embedding = F(0);   // [16,256,64,64]
  const float* image_pe        = F(1);
  const float* point_embedding = F(2);   // [16,64,256]

  // params flattened as sorted-key pytree: final_attn{k,o,q,v}{b,w},
  // final_norm{b,g}, layers[0..1]{i2t, mlp, norm1..4, self_attn, t2i}
  Attn finalAttn = { {F(8), F(7)}, {F(4), F(3)}, {F(10), F(9)}, {F(6), F(5)} };
  LN   finalNorm = { F(12), F(11) };

  struct Layer { Attn self_a, t2i, i2t; Lin lin1, lin2; LN n1, n2, n3, n4; };
  auto mkAttn = [&](int b0) {  // b0 -> x.k.b
    return Attn{ {F(b0 + 5), F(b0 + 4)}, {F(b0 + 1), F(b0 + 0)},
                 {F(b0 + 7), F(b0 + 6)}, {F(b0 + 3), F(b0 + 2)} };
  };
  Layer L[2];
  for (int i = 0; i < 2; ++i) {
    int b0 = 13 + 36 * i;
    L[i].i2t    = mkAttn(b0 + 0);
    L[i].lin1   = { F(b0 + 9),  F(b0 + 8) };
    L[i].lin2   = { F(b0 + 11), F(b0 + 10) };
    L[i].n1     = { F(b0 + 13), F(b0 + 12) };
    L[i].n2     = { F(b0 + 15), F(b0 + 14) };
    L[i].n3     = { F(b0 + 17), F(b0 + 16) };
    L[i].n4     = { F(b0 + 19), F(b0 + 18) };
    L[i].self_a = mkAttn(b0 + 20);
    L[i].t2i    = mkAttn(b0 + 28);
  }

  // Output tuple (queries, keys) lives directly in d_out.
  const size_t QN = (size_t)B_ * NPTS * EMBED;   // 262144
  const size_t KN = (size_t)B_ * HW_ * EMBED;    // 16777216
  float* queries = (float*)d_out;
  float* keys    = queries + QN;

  // Workspace layout (floats)
  float* ws = (float*)d_ws;
  float* key_pe = ws;            ws += KN;
  float* kin    = ws;            ws += KN;
  float* qin    = ws;            ws += QN;
  float* pq     = ws;            ws += (size_t)B_ * HW_ * 128;
  float* pk     = ws;            ws += (size_t)B_ * HW_ * 128;
  float* pv     = ws;            ws += (size_t)B_ * HW_ * 128;
  float* abuf   = ws;            ws += (size_t)B_ * HW_ * 128;
  float* mlpbuf = ws;            ws += (size_t)B_ * NPTS * MLPDIM;

  // keys = permute(image_embedding), key_pe = permute(image_pe)
  bchw_to_bnc<<<8192, 256, 0, stream>>>(image_embedding, keys);
  bchw_to_bnc<<<8192, 256, 0, stream>>>(image_pe, key_pe);
  hipMemcpyAsync(queries, point_embedding, QN * sizeof(float),
                 hipMemcpyDeviceToDevice, stream);

  for (int i = 0; i < 2; ++i) {
    const Layer& lp = L[i];
    // ---- self attention on queries ----
    if (i == 0) {
      run_attention(lp.self_a, queries, queries, queries, NPTS, NPTS, EMBED,
                    nullptr, queries, pq, pk, pv, abuf, stream);
    } else {
      launch_add(queries, point_embedding, qin, QN, stream);
      run_attention(lp.self_a, qin, qin, queries, NPTS, NPTS, EMBED,
                    queries, queries, pq, pk, pv, abuf, stream);
    }
    launch_ln(lp.n1, queries, B_ * NPTS, stream);

    // ---- token -> image cross attention ----
    launch_add(queries, point_embedding, qin, QN, stream);
    launch_add(keys, key_pe, kin, KN, stream);
    run_attention(lp.t2i, qin, kin, keys, NPTS, HW_, EMBED / 2,
                  queries, queries, pq, pk, pv, abuf, stream);
    launch_ln(lp.n2, queries, B_ * NPTS, stream);

    // ---- MLP on queries ----
    launch_gemm(queries, lp.lin1.w, lp.lin1.b, nullptr, mlpbuf,
                B_ * NPTS, MLPDIM, EMBED, true, stream);
    launch_gemm(mlpbuf, lp.lin2.w, lp.lin2.b, queries, queries,
                B_ * NPTS, EMBED, MLPDIM, false, stream);
    launch_ln(lp.n3, queries, B_ * NPTS, stream);

    // ---- image -> token cross attention ----
    launch_add(queries, point_embedding, qin, QN, stream);
    launch_add(keys, key_pe, kin, KN, stream);
    run_attention(lp.i2t, kin, qin, queries, HW_, NPTS, EMBED / 2,
                  keys, keys, pq, pk, pv, abuf, stream);
    launch_ln(lp.n4, keys, B_ * HW_, stream);
  }

  // ---- final token -> image attention ----
  launch_add(queries, point_embedding, qin, QN, stream);
  launch_add(keys, key_pe, kin, KN, stream);
  run_attention(finalAttn, qin, kin, keys, NPTS, HW_, EMBED / 2,
                queries, queries, pq, pk, pv, abuf, stream);
  launch_ln(finalNorm, queries, B_ * NPTS, stream);
}
